// Model_11235634446944
// MI455X (gfx1250) — matrix operand order repaired and checked by reading, not compiled, not run on hardware
//
#include <hip/hip_runtime.h>
#include <hip/hip_bf16.h>

typedef __attribute__((ext_vector_type(16))) _Float16 v16h;
typedef __attribute__((ext_vector_type(8)))  _Float16 v8h;
typedef __attribute__((ext_vector_type(8)))  float    v8f;

#define N_NODES  100000
#define N_EDGES  1600000
#define N_GRAPHS 512
#define F_IN     64
#define H        128
#define STEPS    6
#define ROW_TILES (N_NODES / 16)   // 6250, exact

// ---------------------------------------------------------------------------
// Fast activations (single v_exp_f32 each; gates are saturating so the fast
// path matches the reference to ~1 ulp of f32 in the useful range)
// ---------------------------------------------------------------------------
__device__ inline float fast_sigmoid(float x) { return 1.0f / (1.0f + __expf(-x)); }
__device__ inline float fast_tanh(float x)    { return 2.0f / (1.0f + __expf(-2.0f * x)) - 1.0f; }

// ---------------------------------------------------------------------------
// WMMA fragment helpers (layouts per CDNA5 ISA 7.12.2, wave32)
// ---------------------------------------------------------------------------
__device__ inline v16h combine16(v8h lo, v8h hi) {
    v16h a;
#pragma unroll
    for (int i = 0; i < 8; ++i) { a[i] = lo[i]; a[8 + i] = hi[i]; }
    return a;
}

// A 16x32 from packed-f16 row-major source: two contiguous b128 loads per lane.
// lanes 0-15  : row=lane,    elems 0-7 = K kbase+0..7,  8-15 = K kbase+16..23
// lanes 16-31 : row=lane-16, elems 0-7 = K kbase+8..15, 8-15 = K kbase+24..31
__device__ inline v16h load_A_f16(const _Float16* base, int ld, int row0, int kbase, int lane) {
    int r   = lane & 15;
    int kof = (lane & 16) ? 8 : 0;
    const _Float16* p = base + (size_t)(row0 + r) * ld + kbase + kof;
    v8h lo = *(const v8h*)p;
    v8h hi = *(const v8h*)(p + 16);
    return combine16(lo, hi);
}

// A 16x32 from f32 row-major source (agg / x): loads + cvt in-register.
__device__ inline v16h load_A_f32(const float* base, int ld, int row0, int kbase, int lane) {
    int r   = lane & 15;
    int kof = (lane & 16) ? 8 : 0;
    const float* p = base + (size_t)(row0 + r) * ld + kbase + kof;
    v16h a;
#pragma unroll
    for (int i = 0; i < 8; ++i) a[i]     = (_Float16)p[i];
#pragma unroll
    for (int i = 0; i < 8; ++i) a[8 + i] = (_Float16)p[16 + i];
    return a;
}

// B 32x16 with B[k][n] = Wt[n][k], Wt packed f16 row-major [N x K] (ld = K):
// two runs of 8 contiguous halves per lane -> two b128 loads.
// lanes 0-15: n=n0+lane, elems 0-7 = K kbase+0..7, 8-15 = K kbase+16..23; lanes 16-31: n=n0+lane-16, elems 0-7 = K kbase+8..15, 8-15 = K kbase+24..31
__device__ inline v16h load_B_f16T(const _Float16* Wt, int ld, int kbase, int n0, int lane) {
    int n  = n0 + (lane & 15);
    int k0 = kbase + ((lane & 16) ? 8 : 0);
    const _Float16* p = Wt + (size_t)n * ld + k0;
    v8h lo = *(const v8h*)p;
    v8h hi = *(const v8h*)(p + 16);
    return combine16(lo, hi);
}

// D (f32 16x16): VGPR v -> row v (lanes 0-15, col=lane) / row v+8 (lanes 16-31)
__device__ inline void store_D(float* out, int ld, int row0, int n0, int lane, v8f c) {
    int col   = lane & 15;
    int rbase = (lane & 16) ? 8 : 0;
#pragma unroll
    for (int v = 0; v < 8; ++v)
        out[(size_t)(row0 + rbase + v) * ld + n0 + col] = c[v];
}

// D store with tanh + f16 mirror (input layer)
__device__ inline void store_D_tanh2(float* out, _Float16* out16, int ld, int row0, int n0,
                                     int lane, v8f c) {
    int col   = lane & 15;
    int rbase = (lane & 16) ? 8 : 0;
#pragma unroll
    for (int v = 0; v < 8; ++v) {
        float val = fast_tanh(c[v]);
        size_t g = (size_t)(row0 + rbase + v) * ld + n0 + col;
        out[g]   = val;
        out16[g] = (_Float16)val;
    }
}

__device__ inline void store_D_lds(float* lds, int ld, int n0, int lane, v8f c) {
    int col   = lane & 15;
    int rbase = (lane & 16) ? 8 : 0;
#pragma unroll
    for (int v = 0; v < 8; ++v) lds[(size_t)(rbase + v) * ld + n0 + col] = c[v];
}

__device__ inline v8f wmma_f16(v16h a, v16h b, v8f c) {
    return __builtin_amdgcn_wmma_f32_16x16x32_f16(false, a, false, b, (short)0, c, false, false);
}

// ---------------------------------------------------------------------------
// Weight preconversion: f32 -> packed f16, once per launch (tiny).
// ---------------------------------------------------------------------------
__global__ void convert_transpose(const float* __restrict__ src, _Float16* __restrict__ dst,
                                  int K, int N) {
    // src[K][N] -> dst[N][K]
    int t = blockIdx.x * blockDim.x + threadIdx.x;
    if (t >= K * N) return;
    int n = t / K, k = t - n * K;
    dst[t] = (_Float16)src[(size_t)k * N + n];
}

__global__ void convert_copy(const float* __restrict__ src, _Float16* __restrict__ dst, int count) {
    int t = blockIdx.x * blockDim.x + threadIdx.x;
    if (t < count) dst[t] = (_Float16)src[t];
}

// ---------------------------------------------------------------------------
// Kernel 1: h = tanh(x @ W_in)   [100000 x 64] @ [64 x 128]
// ---------------------------------------------------------------------------
__global__ void input_gemm(const float* __restrict__ x, const _Float16* __restrict__ Wt_in16,
                           float* __restrict__ h, _Float16* __restrict__ h16) {
    int rowTile = blockIdx.x;
    int wave = threadIdx.x >> 5;
    int lane = threadIdx.x & 31;
    int row0 = rowTile * 16;
    int n0   = wave * 16;
    v8f acc = {};
#pragma unroll
    for (int ko = 0; ko < F_IN / 32; ++ko) {
        v16h a = load_A_f32(x, F_IN, row0, ko * 32, lane);
        v16h b = load_B_f16T(Wt_in16, F_IN, ko * 32, n0, lane);  // Wt_in16: [128][64]
        acc = wmma_f16(a, b, acc);
    }
    store_D_tanh2(h, h16, H, row0, n0, lane, acc);
}

// ---------------------------------------------------------------------------
// Kernel 2: m = h @ W_mp[s]   [100000 x 128] @ [128 x 128], all-f16 feeds
// ---------------------------------------------------------------------------
__global__ void msg_gemm(const _Float16* __restrict__ h16, const _Float16* __restrict__ Wt16,
                         float* __restrict__ m) {
    int rowTile = blockIdx.x;
    int wave = threadIdx.x >> 5;
    int lane = threadIdx.x & 31;
    int row0 = rowTile * 16;
    int n0   = wave * 16;
    v8f acc = {};
#pragma unroll
    for (int ko = 0; ko < H / 32; ++ko) {
        v16h a = load_A_f16(h16, H, row0, ko * 32, lane);
        v16h b = load_B_f16T(Wt16, H, ko * 32, n0, lane);  // Wt16: [128][128] = W^T
        acc = wmma_f16(a, b, acc);
    }
    store_D(m, H, row0, n0, lane, acc);
}

// ---------------------------------------------------------------------------
// Kernel 3: agg[dst] += m[src] — wave per edge, float4 per lane, hw f32 atomics
// ---------------------------------------------------------------------------
__global__ void scatter_add(const float* __restrict__ m, const int* __restrict__ src,
                            const int* __restrict__ dst, float* __restrict__ agg) {
    int t = blockIdx.x * blockDim.x + threadIdx.x;
    int e = t >> 5;
    if (e >= N_EDGES) return;
    int lane = t & 31;
    int s = src[e];
    int d = dst[e];
    const float4* mv = (const float4*)(m + (size_t)s * H);
    float4 v = mv[lane];
    float* ap = agg + (size_t)d * H + lane * 4;
    __hip_atomic_fetch_add(ap + 0, v.x, __ATOMIC_RELAXED, __HIP_MEMORY_SCOPE_AGENT);
    __hip_atomic_fetch_add(ap + 1, v.y, __ATOMIC_RELAXED, __HIP_MEMORY_SCOPE_AGENT);
    __hip_atomic_fetch_add(ap + 2, v.z, __ATOMIC_RELAXED, __HIP_MEMORY_SCOPE_AGENT);
    __hip_atomic_fetch_add(ap + 3, v.w, __ATOMIC_RELAXED, __HIP_MEMORY_SCOPE_AGENT);
}

// ---------------------------------------------------------------------------
// Kernel 4: fused GRU. gi = agg @ W_ih^T, gh = h @ W_hh^T staged in LDS,
// then gates + in-place update of h (and its f16 mirror).
// ---------------------------------------------------------------------------
__global__ void gru_fused(const float* __restrict__ agg, float* __restrict__ h,
                          _Float16* __restrict__ h16,
                          const _Float16* __restrict__ Wih16, const _Float16* __restrict__ Whh16,
                          const float* __restrict__ b_ih, const float* __restrict__ b_hh) {
    __shared__ float sgi[16][3 * H];   // 24 KB
    __shared__ float sgh[16][3 * H];   // 24 KB
    int rowTile = blockIdx.x;
    int wave = threadIdx.x >> 5;
    int lane = threadIdx.x & 31;
    int row0 = rowTile * 16;

#pragma unroll
    for (int i = 0; i < 6; ++i) {
        int p = wave * 6 + i;                 // 0..47
        bool is_gi = (p < 24);
        int  tile  = is_gi ? p : (p - 24);    // 0..23 over 384 cols
        const _Float16* W = is_gi ? Wih16 : Whh16; // [384][128] row-major => B = W^T
        v8f acc = {};
#pragma unroll
        for (int ko = 0; ko < H / 32; ++ko) {
            v16h a = is_gi ? load_A_f32(agg, H, row0, ko * 32, lane)
                           : load_A_f16(h16, H, row0, ko * 32, lane);
            v16h b = load_B_f16T(W, H, ko * 32, tile * 16, lane);
            acc = wmma_f16(a, b, acc);
        }
        store_D_lds(is_gi ? &sgi[0][0] : &sgh[0][0], 3 * H, tile * 16, lane, acc);
    }
    __syncthreads();

    // Gates: 16 rows x 128 cols = 2048 values, 8 per thread.
    int t = threadIdx.x;
#pragma unroll
    for (int i = 0; i < 8; ++i) {
        int idx = t * 8 + i;
        int row = idx >> 7;
        int col = idx & (H - 1);
        float i_r = sgi[row][col]         + b_ih[col];
        float i_z = sgi[row][col + H]     + b_ih[col + H];
        float i_n = sgi[row][col + 2 * H] + b_ih[col + 2 * H];
        float h_r = sgh[row][col]         + b_hh[col];
        float h_z = sgh[row][col + H]     + b_hh[col + H];
        float h_n = sgh[row][col + 2 * H] + b_hh[col + 2 * H];
        float r = fast_sigmoid(i_r + h_r);
        float z = fast_sigmoid(i_z + h_z);
        float n = fast_tanh(i_n + r * h_n);
        size_t g = (size_t)(row0 + row) * H + col;
        float hnew = (1.0f - z) * n + z * h[g];
        h[g]   = hnew;
        h16[g] = (_Float16)hnew;
    }
}

// ---------------------------------------------------------------------------
// Kernel 5: per-graph sums + counts (wave per node, float4 per lane)
// ---------------------------------------------------------------------------
__global__ void pool_scatter(const float* __restrict__ h, const int* __restrict__ batch,
                             float* __restrict__ sums, float* __restrict__ cnt) {
    int t = blockIdx.x * blockDim.x + threadIdx.x;
    int node = t >> 5;
    if (node >= N_NODES) return;
    int lane = t & 31;
    int b = batch[node];
    const float4* hv = (const float4*)(h + (size_t)node * H);
    float4 v = hv[lane];
    float* sp = sums + (size_t)b * H + lane * 4;
    __hip_atomic_fetch_add(sp + 0, v.x, __ATOMIC_RELAXED, __HIP_MEMORY_SCOPE_AGENT);
    __hip_atomic_fetch_add(sp + 1, v.y, __ATOMIC_RELAXED, __HIP_MEMORY_SCOPE_AGENT);
    __hip_atomic_fetch_add(sp + 2, v.z, __ATOMIC_RELAXED, __HIP_MEMORY_SCOPE_AGENT);
    __hip_atomic_fetch_add(sp + 3, v.w, __ATOMIC_RELAXED, __HIP_MEMORY_SCOPE_AGENT);
    if (lane == 0)
        __hip_atomic_fetch_add(cnt + b, 1.0f, __ATOMIC_RELAXED, __HIP_MEMORY_SCOPE_AGENT);
}

// ---------------------------------------------------------------------------
// Kernel 6: out[g] = relu(sums/cnt) . W_p + b_p   (block per graph, 128 thr)
// ---------------------------------------------------------------------------
__global__ void pool_finalize(const float* __restrict__ sums, const float* __restrict__ cnt,
                              const float* __restrict__ W_p, const float* __restrict__ b_p,
                              float* __restrict__ out) {
    __shared__ float red[H];
    int g = blockIdx.x;
    int j = threadIdx.x;
    float c = fmaxf(cnt[g], 1.0f);
    float pooled = sums[(size_t)g * H + j] / c;
    red[j] = fmaxf(pooled, 0.0f) * W_p[j];
    __syncthreads();
#pragma unroll
    for (int s = H / 2; s > 0; s >>= 1) {
        if (j < s) red[j] += red[j + s];
        __syncthreads();
    }
    if (j == 0) out[g] = red[0] + b_p[0];
}

// ---------------------------------------------------------------------------
extern "C" void kernel_launch(void* const* d_in, const int* in_sizes, int n_in,
                              void* d_out, int out_size, void* d_ws, size_t ws_size,
                              hipStream_t stream) {
    const float* x     = (const float*)d_in[0];
    const int*   edges = (const int*)d_in[1];   // [2, N_EDGES]: src then dst
    const int*   batch = (const int*)d_in[2];
    const float* W_in  = (const float*)d_in[3];
    const float* W_mp  = (const float*)d_in[4]; // [STEPS, H, H]
    const float* W_ih  = (const float*)d_in[5]; // [3H, H]
    const float* W_hh  = (const float*)d_in[6];
    const float* b_ih  = (const float*)d_in[7];
    const float* b_hh  = (const float*)d_in[8];
    const float* W_p   = (const float*)d_in[9];
    const float* b_p   = (const float*)d_in[10];
    float* out = (float*)d_out;

    const int* src = edges;
    const int* dst = edges + N_EDGES;

    // ---- workspace layout (all sections keep 16B alignment) ----
    float* h    = (float*)d_ws;                      // [N_NODES * H] f32
    float* m    = h   + (size_t)N_NODES * H;         // [N_NODES * H] f32
    float* agg  = m   + (size_t)N_NODES * H;         // [N_NODES * H] f32
    float* sums = agg + (size_t)N_NODES * H;         // [N_GRAPHS * H] f32
    float* cnt  = sums + (size_t)N_GRAPHS * H;       // [N_GRAPHS] f32
    _Float16* h16     = (_Float16*)(cnt + N_GRAPHS);          // [N_NODES * H]
    _Float16* Wt_in16 = h16 + (size_t)N_NODES * H;            // [H][F_IN]  (W_in^T)
    _Float16* Wt_mp16 = Wt_in16 + (size_t)H * F_IN;           // [STEPS][H][H] (W_mp^T)
    _Float16* Wih16   = Wt_mp16 + (size_t)STEPS * H * H;      // [3H][H]
    _Float16* Whh16   = Wih16 + (size_t)3 * H * H;            // [3H][H]

    // ---- preconvert weights to packed f16 (B-fragment friendly layouts) ----
    convert_transpose<<<(H * F_IN + 255) / 256, 256, 0, stream>>>(W_in, Wt_in16, F_IN, H);
    for (int s = 0; s < STEPS; ++s)
        convert_transpose<<<(H * H + 255) / 256, 256, 0, stream>>>(
            W_mp + (size_t)s * H * H, Wt_mp16 + (size_t)s * H * H, H, H);
    convert_copy<<<(3 * H * H + 255) / 256, 256, 0, stream>>>(W_ih, Wih16, 3 * H * H);
    convert_copy<<<(3 * H * H + 255) / 256, 256, 0, stream>>>(W_hh, Whh16, 3 * H * H);

    input_gemm<<<ROW_TILES, 256, 0, stream>>>(x, Wt_in16, h, h16);

    for (int s = 0; s < STEPS; ++s) {
        hipMemsetAsync(agg, 0, (size_t)N_NODES * H * sizeof(float), stream);
        msg_gemm<<<ROW_TILES, 256, 0, stream>>>(h16, Wt_mp16 + (size_t)s * H * H, m);
        scatter_add<<<(N_EDGES * 32) / 256, 256, 0, stream>>>(m, src, dst, agg);
        gru_fused<<<ROW_TILES, 256, 0, stream>>>(agg, h, h16, Wih16, Whh16, b_ih, b_hh);
    }

    hipMemsetAsync(sums, 0, (size_t)(N_GRAPHS * H + N_GRAPHS) * sizeof(float), stream);
    pool_scatter<<<(N_NODES * 32 + 255) / 256, 256, 0, stream>>>(h, batch, sums, cnt);
    pool_finalize<<<N_GRAPHS, H, 0, stream>>>(sums, cnt, W_p, b_p, out);
}
